// SASA_CNN_52767968198946
// MI455X (gfx1250) — matrix operand order repaired and checked by reading, not compiled, not run on hardware
//
#include <hip/hip_runtime.h>
#include <hip/hip_bf16.h>

typedef __bf16 bf16_t;
typedef __attribute__((ext_vector_type(16))) __bf16 v16bf;
typedef __attribute__((ext_vector_type(8)))  __bf16 v8bf;
typedef __attribute__((ext_vector_type(8)))  float  v8f;
typedef unsigned int u32;

// ---------------- WMMA helpers (gfx1250 bf16 16x16x32, f32 accum) ----------------
__device__ __forceinline__ v16bf frag_two_chunks(const bf16_t* c0, const bf16_t* c1) {
  union { v16bf v; v8bf h[2]; } u;
  u.h[0] = *(const v8bf*)c0;
  u.h[1] = *(const v8bf*)c1;
  return u.v;
}
__device__ __forceinline__ v16bf frag_contig16(const bf16_t* p) {
  return *(const v16bf*)p;
}
__device__ __forceinline__ v8f wmma_bf16(v16bf a, v16bf b, v8f c) {
#if defined(__gfx1250__)
  return __builtin_amdgcn_wmma_f32_16x16x32_bf16(false, a, false, b, (short)0, c, false, false);
#else
  return c;
#endif
}

// ---------------- CDNA5 async global->LDS copy (ASYNCcnt path) ----------------
// LDS aperture maps generic addr[31:0] directly to the wave's LDS offset (ISA 10.2),
// so truncating the generic shared pointer gives the dsaddr VGPR value.
__device__ __forceinline__ void async_copy_b128(void* lds_dst, const void* gsrc) {
#if defined(__gfx1250__)
  unsigned l = (unsigned)(unsigned long long)(uintptr_t)lds_dst;
  unsigned long long g = (unsigned long long)(uintptr_t)gsrc;
  asm volatile("global_load_async_to_lds_b128 %0, %1, off"
               :: "v"(l), "v"(g) : "memory");
#else
  (void)lds_dst; (void)gsrc;
#endif
}
__device__ __forceinline__ void wait_async0() {
#if defined(__gfx1250__)
#if __has_builtin(__builtin_amdgcn_s_wait_asynccnt)
  __builtin_amdgcn_s_wait_asynccnt(0);
#else
  asm volatile("s_wait_asynccnt 0x0" ::: "memory");
#endif
#endif
}

// ---------------- model constants ----------------
#define T1    129           // after conv1+pool
#define T2    130           // conv2 output length (pre-pool)
#define T2P   66            // after pool
#define T3    67            // conv3 output length (pre-pool)
#define BN_EPS 1e-5f

// ---------------- weight prep: transpose + bf16 convert ----------------
__global__ void prep_kernel(const float* __restrict__ cw2, const float* __restrict__ cw3,
                            const float* __restrict__ Wq, const float* __restrict__ Wk,
                            const float* __restrict__ Wv, const float* __restrict__ W1,
                            bf16_t* __restrict__ Bt2, bf16_t* __restrict__ Bt3,
                            bf16_t* __restrict__ Wt3, bf16_t* __restrict__ W1t) {
  int idx0 = blockIdx.x * blockDim.x + threadIdx.x;
  int stride = gridDim.x * blockDim.x;
  // cw2 [8][64][32][8] -> Bt2 [8*64][256] with kappa = k*32 + i
  for (int q = idx0; q < 8 * 64 * 256; q += stride) {
    int fo = q >> 8; int rem = q & 255; int k = rem >> 5; int i = rem & 31;
    Bt2[q] = (bf16_t)cw2[(fo * 32 + i) * 8 + k];
  }
  // cw3 [8][128][64][8] -> Bt3 [8*128][512] with kappa = k*64 + i
  for (int q = idx0; q < 8 * 128 * 512; q += stride) {
    int fo = q >> 9; int rem = q & 511; int k = rem >> 6; int i = rem & 63;
    Bt3[q] = (bf16_t)cw3[(fo * 64 + i) * 8 + k];
  }
  // Wq/Wk/Wv [128][128] -> transposed [o][h]
  for (int q = idx0; q < 128 * 128; q += stride) {
    int o = q >> 7; int h = q & 127;
    Wt3[q]         = (bf16_t)Wq[h * 128 + o];
    Wt3[16384 + q] = (bf16_t)Wk[h * 128 + o];
    Wt3[32768 + q] = (bf16_t)Wv[h * 128 + o];
  }
  // W1 [2048][512] -> W1t [512][2048]
  for (int q = idx0; q < 512 * 2048; q += stride) {
    int o = q >> 11; int kp = q & 2047;
    W1t[q] = (bf16_t)W1[kp * 512 + o];
  }
}

// ---------------- conv1 (Cin=1) + BN + ReLU + maxpool -> act1 bf16 [f*256+n][129][32] ----------------
__global__ __launch_bounds__(128) void conv1_kernel(const float* __restrict__ x,
                                                    const float* __restrict__ w1,
                                                    const float* __restrict__ g, const float* __restrict__ be,
                                                    const float* __restrict__ mu, const float* __restrict__ va,
                                                    bf16_t* __restrict__ act1) {
  __shared__ __align__(16) float xp[264];   // input padded by 4 on both ends (len 256)
  __shared__ float wsm[32 * 8];
  __shared__ float scs[32], shs[32];
  int blk = blockIdx.x;              // f*256 + n
  int f = blk >> 8; int n = blk & 255;
  int b = n >> 4, s = n & 15;
  const float* xin = x + (((size_t)b * 8 + f) * 16 + s) * 256;
  int tid = threadIdx.x;
  // zero only the pad regions (disjoint from async-written interior)
  if (tid < 4) { xp[tid] = 0.f; xp[260 + tid] = 0.f; }
  // async copy interior: 256 floats = 64 x b128
  for (int c = tid; c < 64; c += 128)
    async_copy_b128((char*)(xp + 4) + c * 16, (const char*)xin + c * 16);
  for (int i = tid; i < 256; i += 128) wsm[i] = w1[f * 256 + i];
  for (int i = tid; i < 32; i += 128) {
    float sc = g[f * 32 + i] * rsqrtf(va[f * 32 + i] + BN_EPS);
    scs[i] = sc; shs[i] = be[f * 32 + i] - mu[f * 32 + i] * sc;
  }
  wait_async0();
  __syncthreads();
  bf16_t* outp = act1 + (size_t)blk * (T1 * 32);
  for (int idx = tid; idx < T1 * 32; idx += 128) {
    int t1 = idx >> 5; int o = idx & 31;
    const float* w = &wsm[o * 8];
    float sc = scs[o], sh = shs[o];
    float y;
    if (t1 == 0) {
      float a = 0.f;
      for (int k = 0; k < 8; ++k) a += w[k] * xp[k];
      y = fmaxf(a * sc + sh, 0.f);
    } else {
      int ta = 2 * t1 - 1, tb = 2 * t1;
      float a0 = 0.f, a1 = 0.f;
      for (int k = 0; k < 8; ++k) { a0 += w[k] * xp[ta + k]; a1 += w[k] * xp[tb + k]; }
      y = fmaxf(fmaxf(a0 * sc + sh, 0.f), fmaxf(a1 * sc + sh, 0.f));
    }
    outp[idx] = (bf16_t)y;
  }
}

// ---------------- conv2: implicit GEMM via WMMA, + BN + ReLU + pool -> act2 bf16 [blk][66][64] ----------------
#define P2ROWS 152
__global__ __launch_bounds__(128) void conv2_kernel(const bf16_t* __restrict__ act1,
                                                    const bf16_t* __restrict__ Bt2,
                                                    const float* __restrict__ g, const float* __restrict__ be,
                                                    const float* __restrict__ mu, const float* __restrict__ va,
                                                    bf16_t* __restrict__ act2) {
  __shared__ __align__(16) bf16_t P[P2ROWS * 32];   // padded activation slab
  __shared__ __align__(16) bf16_t Y[T2 * 64];       // post-BN/ReLU pre-pool
  int blk = blockIdx.x;            // f*256 + n
  int f = blk >> 8;
  int tid = threadIdx.x;
  // zero pad rows only: rows 0..3 and 133..151 (disjoint from async region rows 4..132)
  for (int i = tid; i < (4 * 32) / 2; i += 128) ((u32*)P)[i] = 0u;
  for (int i = tid; i < (19 * 32) / 2; i += 128) ((u32*)(P + 133 * 32))[i] = 0u;
  // async copy valid slab: 129*32*2 = 8256 B = 516 x b128
  {
    const char* gsrc = (const char*)(act1 + (size_t)blk * (T1 * 32));
    char* ldst = (char*)(P + 4 * 32);
    for (int c = tid; c < 516; c += 128)
      async_copy_b128(ldst + c * 16, gsrc + c * 16);
  }
  wait_async0();
  __syncthreads();
  int wave = tid >> 5, lane = tid & 31;
  int lh = lane >> 4, ln = lane & 15;
  int o = wave * 16 + ln;                  // this lane's output channel
  float sc = g[f * 64 + o] * rsqrtf(va[f * 64 + o] + BN_EPS);
  float sh = be[f * 64 + o] - mu[f * 64 + o] * sc;
  const bf16_t* Bbase = Bt2 + ((size_t)(f * 64 + o)) * 256;
  __builtin_prefetch(Bbase, 0, 1);
  v8f acc[9];
#pragma unroll
  for (int mm = 0; mm < 9; ++mm) acc[mm] = (v8f){0.f, 0.f, 0.f, 0.f, 0.f, 0.f, 0.f, 0.f};
#pragma unroll
  for (int kk = 0; kk < 8; ++kk) {         // K step == tap index (32 channels each)
    v16bf bfg = frag_contig16(Bbase + kk * 32 + lh * 16);   // reused across all M tiles
#pragma unroll
    for (int mm = 0; mm < 9; ++mm) {
      const bf16_t* arow = P + (mm * 16 + ln + kk) * 32 + lh * 16;
      v16bf af = frag_two_chunks(arow, arow + 8);
      acc[mm] = wmma_bf16(af, bfg, acc[mm]);
    }
  }
#pragma unroll
  for (int mm = 0; mm < 9; ++mm)
    for (int r = 0; r < 8; ++r) {
      int t = mm * 16 + lh * 8 + r;
      if (t < T2) Y[t * 64 + o] = (bf16_t)fmaxf(acc[mm][r] * sc + sh, 0.f);
    }
  __syncthreads();
  bf16_t* outp = act2 + (size_t)blk * (T2P * 64);
  for (int idx = tid; idx < T2P * 64; idx += 128) {
    int j = idx >> 6; int c = idx & 63;
    int i0 = 2 * j - 1, i1 = 2 * j;
    float a = (i0 >= 0) ? (float)Y[i0 * 64 + c] : -__builtin_inff();
    float b = (i1 < T2) ? (float)Y[i1 * 64 + c] : -__builtin_inff();
    outp[idx] = (bf16_t)fmaxf(a, b);
  }
}

// ---------------- conv3: implicit GEMM via WMMA, + BN + ReLU + pool + mean -> rep/repb/Kn ----------------
#define P3ROWS 88
__global__ __launch_bounds__(128) void conv3_kernel(const bf16_t* __restrict__ act2,
                                                    const bf16_t* __restrict__ Bt3,
                                                    const float* __restrict__ g, const float* __restrict__ be,
                                                    const float* __restrict__ mu, const float* __restrict__ va,
                                                    float* __restrict__ rep, bf16_t* __restrict__ repb,
                                                    float* __restrict__ Kn) {
  __shared__ __align__(16) bf16_t P[P3ROWS * 64];
  __shared__ __align__(16) bf16_t Y[T3 * 128];
  __shared__ float red[128];
  int blk = blockIdx.x;            // f*256 + n
  int f = blk >> 8; int n = blk & 255;
  int tid = threadIdx.x;
  // zero pad rows: 0..3 and 70..87 (disjoint from async region rows 4..69)
  for (int i = tid; i < (4 * 64) / 2; i += 128) ((u32*)P)[i] = 0u;
  for (int i = tid; i < (18 * 64) / 2; i += 128) ((u32*)(P + 70 * 64))[i] = 0u;
  // async copy valid slab: 66*64*2 = 8448 B = 528 x b128
  {
    const char* gsrc = (const char*)(act2 + (size_t)blk * (T2P * 64));
    char* ldst = (char*)(P + 4 * 64);
    for (int c = tid; c < 528; c += 128)
      async_copy_b128(ldst + c * 16, gsrc + c * 16);
  }
  wait_async0();
  __syncthreads();
  int wave = tid >> 5, lane = tid & 31;
  int lh = lane >> 4, ln = lane & 15;
  for (int nt2 = 0; nt2 < 2; ++nt2) {
    int o = (wave + nt2 * 4) * 16 + ln;
    float sc = g[f * 128 + o] * rsqrtf(va[f * 128 + o] + BN_EPS);
    float sh = be[f * 128 + o] - mu[f * 128 + o] * sc;
    const bf16_t* Bbase = Bt3 + ((size_t)(f * 128 + o)) * 512;
    __builtin_prefetch(Bbase, 0, 1);
    v8f acc[5];
#pragma unroll
    for (int mm = 0; mm < 5; ++mm) acc[mm] = (v8f){0.f, 0.f, 0.f, 0.f, 0.f, 0.f, 0.f, 0.f};
#pragma unroll
    for (int kk = 0; kk < 16; ++kk) {      // tap = kk>>1, channel half = (kk&1)*32
      v16bf bfg = frag_contig16(Bbase + kk * 32 + lh * 16);  // reused across M tiles
#pragma unroll
      for (int mm = 0; mm < 5; ++mm) {
        const bf16_t* arow = P + (mm * 16 + ln + (kk >> 1)) * 64 + (kk & 1) * 32 + lh * 16;
        v16bf af = frag_two_chunks(arow, arow + 8);
        acc[mm] = wmma_bf16(af, bfg, acc[mm]);
      }
    }
#pragma unroll
    for (int mm = 0; mm < 5; ++mm)
      for (int r = 0; r < 8; ++r) {
        int t = mm * 16 + lh * 8 + r;
        if (t < T3) Y[t * 128 + o] = (bf16_t)fmaxf(acc[mm][r] * sc + sh, 0.f);
      }
  }
  __syncthreads();
  // pool (67->34) + mean + row normalize
  int c = tid;
  float s = (float)Y[c];
  for (int j = 1; j < 34; ++j)
    s += fmaxf((float)Y[(2 * j - 1) * 128 + c], (float)Y[(2 * j) * 128 + c]);
  float repv = s * (1.f / 34.f);
  size_t base = ((size_t)f * 256 + n) * 128 + c;
  rep[base] = repv;
  repb[base] = (bf16_t)repv;
  red[c] = repv * repv;
  __syncthreads();
  for (int st = 64; st > 0; st >>= 1) {
    if (tid < st) red[tid] += red[tid + st];
    __syncthreads();
  }
  float nrm = sqrtf(red[0]);
  Kn[base] = repv / fmaxf(nrm, 1e-12f);
}

// ---------------- Q/K/V projection: rep @ W + b, activation (elu / leaky) ----------------
__global__ __launch_bounds__(128) void qkv_kernel(const bf16_t* __restrict__ repb,
                                                  const bf16_t* __restrict__ Wt3,
                                                  const float* __restrict__ bq, const float* __restrict__ bk,
                                                  const float* __restrict__ bv,
                                                  float* __restrict__ Qb, float* __restrict__ Kb,
                                                  float* __restrict__ Vb) {
  int which = blockIdx.y;                     // 0=Q,1=K,2=V
  const bf16_t* W = Wt3 + (size_t)which * 16384;
  const float* bias = (which == 0) ? bq : (which == 1) ? bk : bv;
  float* out = (which == 0) ? Qb : (which == 1) ? Kb : Vb;
  int blk = blockIdx.x;                       // f*16 + mtile
  int f = blk >> 4; int m0 = (blk & 15) * 16;
  int tid = threadIdx.x;
  int wave = tid >> 5, lane = tid & 31;
  int lh = lane >> 4, ln = lane & 15;
  const bf16_t* Abase = repb + ((size_t)f * 256 + m0 + ln) * 128;
  for (int nt2 = 0; nt2 < 2; ++nt2) {
    int oo = (wave + nt2 * 4) * 16 + ln;
    const bf16_t* Bbase = W + (size_t)oo * 128;
    v8f acc = {0.f, 0.f, 0.f, 0.f, 0.f, 0.f, 0.f, 0.f};
#pragma unroll
    for (int kk = 0; kk < 4; ++kk) {
      const bf16_t* arow = Abase + kk * 32 + lh * 16;
      v16bf af = frag_two_chunks(arow, arow + 8);
      v16bf bfg = frag_contig16(Bbase + kk * 32 + lh * 16);
      acc = wmma_bf16(af, bfg, acc);
    }
    float bb = bias[oo];
    for (int r = 0; r < 8; ++r) {
      int row = m0 + lh * 8 + r;
      float v = acc[r] + bb;
      if (which == 0) v = (v > 0.f) ? v : expm1f(v);      // elu
      else            v = (v > 0.f) ? v : 0.01f * v;       // leaky_relu
      out[((size_t)f * 256 + row) * 128 + oo] = v;
    }
  }
}

// ---------------- sparsemax over 16 (single thread) ----------------
__device__ void sparsemax16(const float* z, float* p) {
  float zs[16];
  for (int i = 0; i < 16; ++i) zs[i] = z[i];
  for (int i = 1; i < 16; ++i) {           // insertion sort, descending
    float v = zs[i]; int j = i - 1;
    while (j >= 0 && zs[j] < v) { zs[j + 1] = zs[j]; --j; }
    zs[j + 1] = v;
  }
  float cum = 0.f, cumk = 1.f; int ksup = 1;
  for (int k = 1; k <= 16; ++k) {
    cum += zs[k - 1];
    if (1.f + (float)k * zs[k - 1] > cum) { ksup = k; cumk = cum; }
  }
  float tau = (cumk - 1.f) / (float)ksup;
  for (int i = 0; i < 16; ++i) p[i] = fmaxf(z[i] - tau, 0.f);
}

// ---------------- intra attention: scores -> sparsemax -> Z (normalized) ----------------
__global__ __launch_bounds__(128) void intra_kernel(const float* __restrict__ Q,
                                                    const float* __restrict__ K,
                                                    const float* __restrict__ V,
                                                    float* __restrict__ aw_out,
                                                    float* __restrict__ Z) {
  int f = blockIdx.x, b = blockIdx.y; int tid = threadIdx.x;
  __shared__ float mk[128], sc16[16], aw[16], red[128];
  const float* Kb = K + ((size_t)f * 256 + b * 16) * 128;
  float m = 0.f;
  for (int s = 0; s < 16; ++s) m += Kb[s * 128 + tid];
  mk[tid] = m * (1.f / 16.f);
  __syncthreads();
  if (tid < 16) {
    const float* Qr = Q + ((size_t)f * 256 + b * 16 + tid) * 128;
    float d = 0.f;
    for (int h = 0; h < 128; ++h) d += Qr[h] * mk[h];
    sc16[tid] = d * 0.08838834764831845f;   // 1/sqrt(128)
  }
  __syncthreads();
  if (tid == 0) sparsemax16(sc16, aw);
  __syncthreads();
  if (tid < 16) aw_out[(f * 16 + b) * 16 + tid] = aw[tid];
  const float* Vb = V + ((size_t)f * 256 + b * 16) * 128;
  float z = 0.f;
  for (int s = 0; s < 16; ++s) z += aw[s] * Vb[s * 128 + tid];
  red[tid] = z * z;
  __syncthreads();
  for (int st = 64; st > 0; st >>= 1) {
    if (tid < st) red[tid] += red[tid + st];
    __syncthreads();
  }
  float nrm = sqrtf(red[0]);
  Z[((size_t)f * 16 + b) * 128 + tid] = z / fmaxf(nrm, 1e-12f);
}

// ---------------- inter attention: cosine scores -> sparsemax ----------------
__global__ __launch_bounds__(32) void inter_kernel(const float* __restrict__ Z,
                                                   const float* __restrict__ Kn,
                                                   float* __restrict__ aw_out) {
  int i = blockIdx.x, f = blockIdx.y, b = blockIdx.z; int tid = threadIdx.x;
  __shared__ float sc16[16], aw[16];
  if (tid < 16) {
    const float* Zr = Z + ((size_t)i * 16 + b) * 128;
    const float* Kr = Kn + ((size_t)f * 256 + b * 16 + tid) * 128;
    float d = 0.f;
    for (int h = 0; h < 128; ++h) d += Zr[h] * Kr[h];
    sc16[tid] = d;
  }
  __syncthreads();
  if (tid == 0) sparsemax16(sc16, aw);
  __syncthreads();
  if (tid < 16) aw_out[(((size_t)i * 8 + f) * 16 + b) * 16 + tid] = aw[tid];
}

// ---------------- U + Hc normalize -> feat [16][2048] ----------------
__global__ __launch_bounds__(128) void ufeat_kernel(const float* __restrict__ Z,
                                                    const float* __restrict__ rep,
                                                    const float* __restrict__ aw2,
                                                    float* __restrict__ feat) {
  int i = blockIdx.x, b = blockIdx.y; int tid = threadIdx.x;
  __shared__ float red[128];
  float u = 0.f;
  for (int f = 0; f < 8; ++f) {
    const float* a = aw2 + (((size_t)i * 8 + f) * 16 + b) * 16;
    const float* r = rep + ((size_t)f * 256 + b * 16) * 128 + tid;
    for (int s = 0; s < 16; ++s) u += a[s] * r[s * 128];
  }
  u *= 0.125f;
  float z = Z[((size_t)i * 16 + b) * 128 + tid];
  red[tid] = z * z + u * u;
  __syncthreads();
  for (int st = 64; st > 0; st >>= 1) {
    if (tid < st) red[tid] += red[tid + st];
    __syncthreads();
  }
  float inv = 1.f / fmaxf(sqrtf(red[0]), 1e-12f);
  feat[(size_t)b * 2048 + i * 256 + tid] = z * inv;
  feat[(size_t)b * 2048 + i * 256 + 128 + tid] = u * inv;
}

// ---------------- BN(feat) -> bf16 ----------------
__global__ void bnfeat_kernel(const float* __restrict__ feat,
                              const float* __restrict__ g, const float* __restrict__ be,
                              const float* __restrict__ mu, const float* __restrict__ va,
                              bf16_t* __restrict__ featb) {
  int idx = blockIdx.x * blockDim.x + threadIdx.x;
  if (idx < 16 * 2048) {
    int c = idx & 2047;
    float sc = g[c] * rsqrtf(va[c] + BN_EPS);
    featb[idx] = (bf16_t)(sc * (feat[idx] - mu[c]) + be[c]);
  }
}

// ---------------- GEMM1: [16,2048] x [2048,512] via WMMA -> h1 ----------------
__global__ __launch_bounds__(128) void gemm1_kernel(const bf16_t* __restrict__ featb,
                                                    const bf16_t* __restrict__ W1t,
                                                    const float* __restrict__ b1,
                                                    float* __restrict__ h1) {
  int tid = threadIdx.x;
  int wave = tid >> 5, lane = tid & 31;
  int lh = lane >> 4, ln = lane & 15;
  int o = (blockIdx.x * 4 + wave) * 16 + ln;
  const bf16_t* Abase = featb + (size_t)ln * 2048;
  const bf16_t* Bbase = W1t + (size_t)o * 2048;
  __builtin_prefetch(Bbase, 0, 1);
  v8f acc = {0.f, 0.f, 0.f, 0.f, 0.f, 0.f, 0.f, 0.f};
#pragma unroll 8
  for (int kk = 0; kk < 64; ++kk) {
    const bf16_t* arow = Abase + kk * 32 + lh * 16;
    v16bf af = frag_two_chunks(arow, arow + 8);
    v16bf bfg = frag_contig16(Bbase + kk * 32 + lh * 16);
    acc = wmma_bf16(af, bfg, acc);
  }
  float bb = b1[o];
  for (int r = 0; r < 8; ++r) {
    int row = lh * 8 + r;
    h1[(size_t)row * 512 + o] = acc[r] + bb;
  }
}

// ---------------- head: BN + leaky -> GEMM2 -> softmax -> y_pred, CE ----------------
__global__ __launch_bounds__(256) void head_kernel(const float* __restrict__ h1,
                                                   const float* __restrict__ g2, const float* __restrict__ b2g,
                                                   const float* __restrict__ m2, const float* __restrict__ v2,
                                                   const float* __restrict__ W2, const float* __restrict__ bias2,
                                                   const int* __restrict__ srcy,
                                                   float* __restrict__ out, float* __restrict__ ce_out) {
  __shared__ float h2[16 * 512];
  __shared__ float yp[16 * 10];
  __shared__ float cered[16];
  int tid = threadIdx.x;
  for (int idx = tid; idx < 16 * 512; idx += 256) {
    int c = idx & 511;
    float sc = g2[c] * rsqrtf(v2[c] + BN_EPS);
    float v = sc * (h1[idx] - m2[c]) + b2g[c];
    h2[idx] = (v > 0.f) ? v : 0.01f * v;
  }
  __syncthreads();
  if (tid < 160) {
    int b = tid / 10, c = tid % 10;
    float a = bias2[c];
    const float* hr = &h2[b * 512];
    for (int j = 0; j < 512; ++j) a += hr[j] * W2[j * 10 + c];
    yp[tid] = a;
  }
  __syncthreads();
  if (tid < 16) {
    float mx = -__builtin_inff();
    for (int c = 0; c < 10; ++c) mx = fmaxf(mx, yp[tid * 10 + c]);
    float sm = 0.f, e[10];
    for (int c = 0; c < 10; ++c) { e[c] = expf(yp[tid * 10 + c] - mx); sm += e[c]; }
    for (int c = 0; c < 10; ++c) {
      float p = e[c] / sm;
      yp[tid * 10 + c] = p;
      out[tid * 10 + c] = p;
    }
    // CE on already-softmaxed preds (log_softmax of probabilities)
    float mx2 = -__builtin_inff();
    for (int c = 0; c < 10; ++c) mx2 = fmaxf(mx2, yp[tid * 10 + c]);
    float sm2 = 0.f;
    for (int c = 0; c < 10; ++c) sm2 += expf(yp[tid * 10 + c] - mx2);
    float lse = logf(sm2) + mx2;
    int lab = srcy[tid];
    cered[tid] = -(yp[tid * 10 + lab] - lse);
  }
  __syncthreads();
  if (tid == 0) {
    float s = 0.f;
    for (int b = 0; b < 16; ++b) s += cered[b];
    ce_out[0] = s * (1.f / 16.f);
  }
}

// ---------------- MMD domain losses + final scalar ----------------
__global__ __launch_bounds__(256) void loss_kernel(const float* __restrict__ sa, const float* __restrict__ ta,
                                                   const float* __restrict__ sb, const float* __restrict__ tb,
                                                   const float* __restrict__ ce, float* __restrict__ out) {
  __shared__ float da[128];    // [f][s]
  __shared__ float db[1024];   // [i][f][s]
  __shared__ float r8[8];
  __shared__ float lab[2];
  int tid = threadIdx.x;
  if (tid < 128) {
    int f = tid >> 4, s = tid & 15;
    float acc = 0.f;
    for (int b = 0; b < 16; ++b) { int idx = (f * 16 + b) * 16 + s; acc += sa[idx] - ta[idx]; }
    da[tid] = acc * (1.f / 16.f);
  }
  for (int q = tid; q < 1024; q += 256) {
    int i = q >> 7, f = (q >> 4) & 7, s = q & 15;
    float acc = 0.f;
    for (int b = 0; b < 16; ++b) {
      size_t idx = (((size_t)i * 8 + f) * 16 + b) * 16 + s;
      acc += sb[idx] - tb[idx];
    }
    db[q] = acc * (1.f / 16.f);
  }
  __syncthreads();
  if (tid < 8) {
    float s = 0.f;
    for (int j = 0; j < 16; ++j) { float v = da[tid * 16 + j]; s += v * v; }
    r8[tid] = sqrtf(s);
  }
  __syncthreads();
  if (tid == 0) {
    float la = 0.f;
    for (int f = 0; f < 8; ++f) la += r8[f];
    lab[0] = la * 0.125f;
  }
  __syncthreads();
  if (tid < 8) {
    float s = 0.f;
    for (int j = 0; j < 128; ++j) { float v = db[tid * 128 + j]; s += v * v; }
    r8[tid] = sqrtf(s);
  }
  __syncthreads();
  if (tid == 0) {
    float lb = 0.f;
    for (int i = 0; i < 8; ++i) lb += r8[i];
    lb *= 0.125f;
    out[160] = ce[0] + 0.1f * lab[0] + 0.1f * lb;
  }
}

// ---------------- workspace layout ----------------
static constexpr size_t AL(size_t x) { return (x + 4095) & ~size_t(4095); }
constexpr size_t OFF_BT2  = 0;
constexpr size_t OFF_BT3  = AL(OFF_BT2 + 8ull * 64 * 256 * 2);
constexpr size_t OFF_WT3  = AL(OFF_BT3 + 8ull * 128 * 512 * 2);
constexpr size_t OFF_W1T  = AL(OFF_WT3 + 3ull * 128 * 128 * 2);
constexpr size_t OFF_ACT1 = AL(OFF_W1T + 512ull * 2048 * 2);
constexpr size_t OFF_ACT2 = AL(OFF_ACT1 + 2048ull * T1 * 32 * 2);
constexpr size_t OFF_REP  = AL(OFF_ACT2 + 2048ull * T2P * 64 * 2);
constexpr size_t OFF_REPB = AL(OFF_REP + 8ull * 256 * 128 * 4);
constexpr size_t OFF_KN   = AL(OFF_REPB + 8ull * 256 * 128 * 2);
constexpr size_t OFF_Q    = AL(OFF_KN + 8ull * 256 * 128 * 4);
constexpr size_t OFF_K    = AL(OFF_Q + 8ull * 256 * 128 * 4);
constexpr size_t OFF_V    = AL(OFF_K + 8ull * 256 * 128 * 4);
constexpr size_t OFF_Z    = AL(OFF_V + 8ull * 256 * 128 * 4);
constexpr size_t OFF_SA   = AL(OFF_Z + 8ull * 16 * 128 * 4);
constexpr size_t OFF_TA   = AL(OFF_SA + 8ull * 16 * 16 * 4);
constexpr size_t OFF_SB   = AL(OFF_TA + 8ull * 16 * 16 * 4);
constexpr size_t OFF_TB   = AL(OFF_SB + 8ull * 8 * 16 * 16 * 4);
constexpr size_t OFF_FEAT = AL(OFF_TB + 8ull * 8 * 16 * 16 * 4);
constexpr size_t OFF_FEATB= AL(OFF_FEAT + 16ull * 2048 * 4);
constexpr size_t OFF_H1   = AL(OFF_FEATB + 16ull * 2048 * 2);
constexpr size_t OFF_CE   = AL(OFF_H1 + 16ull * 512 * 4);

extern "C" void kernel_launch(void* const* d_in, const int* in_sizes, int n_in,
                              void* d_out, int out_size, void* d_ws, size_t ws_size,
                              hipStream_t stream) {
  const float* src_x = (const float*)d_in[0];
  const int*   src_y = (const int*)d_in[1];
  const float* tgt_x = (const float*)d_in[2];
  const float* cw1   = (const float*)d_in[3];
  const float* bn1_g = (const float*)d_in[4];
  const float* bn1_b = (const float*)d_in[5];
  const float* bn1_m = (const float*)d_in[6];
  const float* bn1_v = (const float*)d_in[7];
  const float* cw2   = (const float*)d_in[8];
  const float* bn2_g = (const float*)d_in[9];
  const float* bn2_b = (const float*)d_in[10];
  const float* bn2_m = (const float*)d_in[11];
  const float* bn2_v = (const float*)d_in[12];
  const float* cw3   = (const float*)d_in[13];
  const float* bn3_g = (const float*)d_in[14];
  const float* bn3_b = (const float*)d_in[15];
  const float* bn3_m = (const float*)d_in[16];
  const float* bn3_v = (const float*)d_in[17];
  const float* Wq    = (const float*)d_in[18];
  const float* bq    = (const float*)d_in[19];
  const float* Wk    = (const float*)d_in[20];
  const float* bk    = (const float*)d_in[21];
  const float* Wv    = (const float*)d_in[22];
  const float* bv    = (const float*)d_in[23];
  const float* bnc1_g = (const float*)d_in[24];
  const float* bnc1_b = (const float*)d_in[25];
  const float* bnc1_m = (const float*)d_in[26];
  const float* bnc1_v = (const float*)d_in[27];
  const float* W1    = (const float*)d_in[28];
  const float* b1    = (const float*)d_in[29];
  const float* bnc2_g = (const float*)d_in[30];
  const float* bnc2_b = (const float*)d_in[31];
  const float* bnc2_m = (const float*)d_in[32];
  const float* bnc2_v = (const float*)d_in[33];
  const float* W2    = (const float*)d_in[34];
  const float* b2    = (const float*)d_in[35];

  char* ws = (char*)d_ws;
  bf16_t* Bt2   = (bf16_t*)(ws + OFF_BT2);
  bf16_t* Bt3   = (bf16_t*)(ws + OFF_BT3);
  bf16_t* Wt3   = (bf16_t*)(ws + OFF_WT3);
  bf16_t* W1t   = (bf16_t*)(ws + OFF_W1T);
  bf16_t* act1  = (bf16_t*)(ws + OFF_ACT1);
  bf16_t* act2  = (bf16_t*)(ws + OFF_ACT2);
  float*  rep   = (float*)(ws + OFF_REP);
  bf16_t* repb  = (bf16_t*)(ws + OFF_REPB);
  float*  Kn    = (float*)(ws + OFF_KN);
  float*  Qb    = (float*)(ws + OFF_Q);
  float*  Kb    = (float*)(ws + OFF_K);
  float*  Vb    = (float*)(ws + OFF_V);
  float*  Zb    = (float*)(ws + OFF_Z);
  float*  sa    = (float*)(ws + OFF_SA);
  float*  ta    = (float*)(ws + OFF_TA);
  float*  sb    = (float*)(ws + OFF_SB);
  float*  tb    = (float*)(ws + OFF_TB);
  float*  feat  = (float*)(ws + OFF_FEAT);
  bf16_t* featb = (bf16_t*)(ws + OFF_FEATB);
  float*  h1    = (float*)(ws + OFF_H1);
  float*  ce    = (float*)(ws + OFF_CE);

  prep_kernel<<<512, 256, 0, stream>>>(cw2, cw3, Wq, Wk, Wv, W1, Bt2, Bt3, Wt3, W1t);

  for (int dom = 0; dom < 2; ++dom) {
    const float* x  = dom ? tgt_x : src_x;
    float* aw1 = dom ? ta : sa;
    float* aw2 = dom ? tb : sb;
    conv1_kernel<<<2048, 128, 0, stream>>>(x, cw1, bn1_g, bn1_b, bn1_m, bn1_v, act1);
    conv2_kernel<<<2048, 128, 0, stream>>>(act1, Bt2, bn2_g, bn2_b, bn2_m, bn2_v, act2);
    conv3_kernel<<<2048, 128, 0, stream>>>(act2, Bt3, bn3_g, bn3_b, bn3_m, bn3_v, rep, repb, Kn);
    qkv_kernel<<<dim3(128, 3), 128, 0, stream>>>(repb, Wt3, bq, bk, bv, Qb, Kb, Vb);
    intra_kernel<<<dim3(8, 16), 128, 0, stream>>>(Qb, Kb, Vb, aw1, Zb);
    inter_kernel<<<dim3(8, 8, 16), 32, 0, stream>>>(Zb, Kn, aw2);
    if (dom == 0)
      ufeat_kernel<<<dim3(8, 16), 128, 0, stream>>>(Zb, rep, aw2, feat);
  }

  bnfeat_kernel<<<128, 256, 0, stream>>>(feat, bnc1_g, bnc1_b, bnc1_m, bnc1_v, featb);
  gemm1_kernel<<<8, 128, 0, stream>>>(featb, W1t, b1, h1);
  head_kernel<<<1, 256, 0, stream>>>(h1, bnc2_g, bnc2_b, bnc2_m, bnc2_v, W2, b2, src_y,
                                     (float*)d_out, ce);
  loss_kernel<<<1, 256, 0, stream>>>(sa, ta, sb, tb, ce, (float*)d_out);
}
